// TurboQuantLinear_25915832664794
// MI455X (gfx1250) — matrix operand order repaired and checked by reading, not compiled, not run on hardware
//
#include <hip/hip_runtime.h>

// ---------------------------------------------------------------------------
// TurboQuantLinear forward for MI455X (gfx1250, wave32, WMMA)
//   out[b,n] = sum_k xrot[b,k] * W'[n,k]
//   xrot[:,g] = x[:,g] @ R_g^T          (per-group 128x128 rotation)
//   W'[n,k]   = codebook[idx[n,k]] * norms[n,k/128] / sqrt(128)
// Intermediates bf16, f32 accumulation via v_wmma_f32_16x16x32_bf16.
// GEMM K-panels staged with GLOBAL_LOAD_ASYNC_TO_LDS_B128 (double-buffered)
// when the toolchain exposes the builtin; synchronous staging otherwise.
// ---------------------------------------------------------------------------

typedef __attribute__((ext_vector_type(16))) __bf16 v16bf;
typedef __attribute__((ext_vector_type(8)))  __bf16 v8bf;
typedef __attribute__((ext_vector_type(8)))  float  v8f;
typedef __attribute__((ext_vector_type(4)))  float  v4f;
typedef __attribute__((ext_vector_type(4)))  int    v4i;

#if defined(__AMDGCN__) && __has_builtin(__builtin_amdgcn_global_load_async_to_lds_b128)
#define USE_ASYNC_LDS 1
#else
#define USE_ASYNC_LDS 0
#endif

namespace {
constexpr int   BDIM = 8192;
constexpr int   KDIM = 4096;
constexpr int   NDIM = 4096;
constexpr int   GRP  = 128;
constexpr int   NG   = KDIM / GRP;          // 32 groups
constexpr float INV_SQRT_G = 0.08838834764831845f;  // 1/sqrt(128)

union Frag {
  v16bf v;
  v8bf  h[2];
};
}  // namespace

#if USE_ASYNC_LDS
#define AS_GLOBAL __attribute__((address_space(1)))
#define AS_LOCAL  __attribute__((address_space(3)))
__device__ inline void async_copy_b128(const void* gsrc, void* ldst) {
  __builtin_amdgcn_global_load_async_to_lds_b128(
      (AS_GLOBAL v4i*)gsrc, (AS_LOCAL v4i*)ldst, 0, 0);
}
__device__ inline void wait_async0() {
#if __has_builtin(__builtin_amdgcn_s_wait_asynccnt)
  __builtin_amdgcn_s_wait_asynccnt(0);
#else
  asm volatile("s_wait_asynccnt 0x0" ::: "memory");
#endif
}
#endif

// ---------------------------------------------------------------------------
// Kernel 1: dequantize weights -> bf16 W' with norms and 1/sqrt(G) folded in.
// ---------------------------------------------------------------------------
__global__ __launch_bounds__(256) void dequant_w_kernel(
    const int* __restrict__ idx, const float* __restrict__ codebook,
    const float* __restrict__ norms, __bf16* __restrict__ Wq) {
  __shared__ float cb[16];
  if (threadIdx.x < 16) cb[threadIdx.x] = codebook[threadIdx.x];
  __syncthreads();

  const long e = ((long)blockIdx.x * 256 + threadIdx.x) * 8;
  const int  n = (int)(e >> 12);            // / KDIM
  const int  k = (int)(e & (KDIM - 1));
  const float s = norms[n * NG + (k >> 7)] * INV_SQRT_G;

  const v4i i0 = *(const v4i*)(idx + e);
  const v4i i1 = *(const v4i*)(idx + e + 4);
  v8bf o;
  o[0] = (__bf16)(cb[i0.x] * s);
  o[1] = (__bf16)(cb[i0.y] * s);
  o[2] = (__bf16)(cb[i0.z] * s);
  o[3] = (__bf16)(cb[i0.w] * s);
  o[4] = (__bf16)(cb[i1.x] * s);
  o[5] = (__bf16)(cb[i1.y] * s);
  o[6] = (__bf16)(cb[i1.z] * s);
  o[7] = (__bf16)(cb[i1.w] * s);
  *(v8bf*)(Wq + e) = o;
}

// ---------------------------------------------------------------------------
// Kernel 2: per-group rotation of x, f32 -> bf16, via WMMA.
// Block = 256 threads (8 waves): 128 x-rows x one group (K=N=128).
// B fragments preloaded per k-step so WMMAs can issue back-to-back.
// ---------------------------------------------------------------------------
__global__ __launch_bounds__(256) void rotate_x_kernel(
    const float* __restrict__ x, const float* __restrict__ rot,
    __bf16* __restrict__ Xr) {
  constexpr int LDR = GRP + 8;  // 136 halfs = 272 B (16B-aligned rows)
  extern __shared__ __bf16 smem_rot[];
  __bf16* Rl = smem_rot;              // [128][LDR]
  __bf16* Xl = smem_rot + GRP * LDR;  // [128][LDR]

  const int g = blockIdx.y;
  const int rowbase = blockIdx.x * 128;
  const int tid = threadIdx.x;

  const float* Rg = rot + (size_t)g * GRP * GRP;
#pragma unroll
  for (int i = 0; i < 16; ++i) {
    const int q = tid + i * 256;     // float4 index
    const int r = q >> 5;
    const int c = (q & 31) << 2;
    const v4f v = ((const v4f*)Rg)[q];
    __bf16* d = Rl + r * LDR + c;
    d[0] = (__bf16)v.x; d[1] = (__bf16)v.y; d[2] = (__bf16)v.z; d[3] = (__bf16)v.w;
  }
  const float* xg = x + (size_t)rowbase * KDIM + g * GRP;
#pragma unroll
  for (int i = 0; i < 16; ++i) {
    const int q = tid + i * 256;
    const int r = q >> 5;
    const int c = (q & 31) << 2;
    const v4f v = *(const v4f*)(xg + (size_t)r * KDIM + c);
    __bf16* d = Xl + r * LDR + c;
    d[0] = (__bf16)v.x; d[1] = (__bf16)v.y; d[2] = (__bf16)v.z; d[3] = (__bf16)v.w;
  }
  __syncthreads();

  const int wave = tid >> 5;
  const int lane = tid & 31;
  const int lh   = lane & 15;
  const int hi   = lane >> 4;
  const int asel = hi << 3;       // A: +8 halfs for lanes 16..31
  const int bsel = hi << 3;       // B: +8 halfs for lanes 16..31
  const int mrow = wave * 16 + lh;

  v8f acc[8];
  const v8f z = {};
#pragma unroll
  for (int t = 0; t < 8; ++t) acc[t] = z;

#pragma unroll
  for (int kk = 0; kk < GRP; kk += 32) {
    Frag a;
    const __bf16* ap = Xl + mrow * LDR + kk + asel;
    a.h[0] = *(const v8bf*)ap;
    a.h[1] = *(const v8bf*)(ap + 16);
    Frag b[8];
#pragma unroll
    for (int t = 0; t < 8; ++t) {
      const __bf16* bp = Rl + (t * 16 + lh) * LDR + kk + bsel;
      b[t].h[0] = *(const v8bf*)bp;
      b[t].h[1] = *(const v8bf*)(bp + 16);
    }
#pragma unroll
    for (int t = 0; t < 8; ++t)
      acc[t] = __builtin_amdgcn_wmma_f32_16x16x32_bf16(
          false, a.v, false, b[t].v, (short)0, acc[t], false, false);
  }

  const int orow  = rowbase + wave * 16 + (hi << 3);
  const int ocolb = g * GRP + lh;
#pragma unroll
  for (int t = 0; t < 8; ++t) {
    __bf16* op = Xr + (size_t)orow * KDIM + ocolb + t * 16;
#pragma unroll
    for (int vv = 0; vv < 8; ++vv) op[(size_t)vv * KDIM] = (__bf16)acc[t][vv];
  }
}

// ---------------------------------------------------------------------------
// Kernel 3: out(B x N, f32) = Xr(B x K, bf16) @ Wq(N x K, bf16)^T
// 128x128 tile, 8 waves (each 32x64). K staged in 64-wide panels,
// double-buffered; async global->LDS copies overlap panel p+1 loads with
// panel p WMMAs. Dynamic LDS: 2 bufs x (A+B) x 128 x 72 halfs = 73728 B.
// ---------------------------------------------------------------------------
__global__ __launch_bounds__(256) void gemm_kernel(
    const __bf16* __restrict__ Xr, const __bf16* __restrict__ Wq,
    float* __restrict__ out) {
  constexpr int LDT   = 72;          // padded stride (halfs); 144 B rows
  constexpr int PANEL = 128 * LDT;   // halfs per matrix panel
  extern __shared__ __bf16 smem_gemm[];  // [2][2][PANEL]

  const int rowbase = blockIdx.x * 128;  // M (batch rows)
  const int colbase = blockIdx.y * 128;  // N (output features)
  const int tid  = threadIdx.x;
  const int wave = tid >> 5;
  const int lane = tid & 31;
  const int lh   = lane & 15;
  const int hi   = lane >> 4;
  const int asel = hi << 3;
  const int bsel = hi << 3;
  const int wm   = wave >> 1;  // 0..3 -> 32-row strip
  const int wn   = wave & 1;   // 0..1 -> 64-col strip

  // Per-thread staging coordinates (4 b128 chunks per matrix per panel).
  const int r0 = tid >> 3;            // 0..31, +32 per step
  const int c0 = (tid & 7) << 3;      // 0,8,..,56 halfs

  v8f acc[2][4];
  const v8f z = {};
#pragma unroll
  for (int i = 0; i < 2; ++i)
#pragma unroll
    for (int j = 0; j < 4; ++j) acc[i][j] = z;

  auto stage_panel = [&](int buf, int ks) {
    __bf16* A = smem_gemm + (size_t)buf * 2 * PANEL;
    __bf16* B = A + PANEL;
#pragma unroll
    for (int i = 0; i < 4; ++i) {
      const int r = r0 + i * 32;
#if USE_ASYNC_LDS
      async_copy_b128(Xr + (size_t)(rowbase + r) * KDIM + ks + c0,
                      A + r * LDT + c0);
      async_copy_b128(Wq + (size_t)(colbase + r) * KDIM + ks + c0,
                      B + r * LDT + c0);
#else
      *(v8bf*)(A + r * LDT + c0) =
          *(const v8bf*)(Xr + (size_t)(rowbase + r) * KDIM + ks + c0);
      *(v8bf*)(B + r * LDT + c0) =
          *(const v8bf*)(Wq + (size_t)(colbase + r) * KDIM + ks + c0);
#endif
    }
  };

  constexpr int NKP = KDIM / 64;  // 64 K-panels
  stage_panel(0, 0);

  for (int p = 0; p < NKP; ++p) {
#if USE_ASYNC_LDS
    wait_async0();        // panel p copies (this thread) complete
#endif
    __syncthreads();      // panel p visible to all waves

    if (p + 1 < NKP) stage_panel((p + 1) & 1, (p + 1) * 64);  // overlap

    const __bf16* A = smem_gemm + (size_t)(p & 1) * 2 * PANEL;
    const __bf16* B = A + PANEL;

#pragma unroll
    for (int kk = 0; kk < 64; kk += 32) {
      Frag a[2], b[4];
#pragma unroll
      for (int i = 0; i < 2; ++i) {
        const __bf16* ap = A + (wm * 32 + i * 16 + lh) * LDT + kk + asel;
        a[i].h[0] = *(const v8bf*)ap;
        a[i].h[1] = *(const v8bf*)(ap + 16);
      }
#pragma unroll
      for (int j = 0; j < 4; ++j) {
        const __bf16* bp = B + (wn * 64 + j * 16 + lh) * LDT + kk + bsel;
        b[j].h[0] = *(const v8bf*)bp;
        b[j].h[1] = *(const v8bf*)(bp + 16);
      }
#pragma unroll
      for (int i = 0; i < 2; ++i)
#pragma unroll
        for (int j = 0; j < 4; ++j)
          acc[i][j] = __builtin_amdgcn_wmma_f32_16x16x32_bf16(
              false, a[i].v, false, b[j].v, (short)0, acc[i][j], false, false);
    }
    __syncthreads();      // all waves done reading buf before its reuse
  }

  // Writeback f32: D layout: VGPR vv -> M = vv + 8*hi, N = lane%16.
#pragma unroll
  for (int i = 0; i < 2; ++i) {
    const int orow = rowbase + wm * 32 + i * 16 + (hi << 3);
#pragma unroll
    for (int j = 0; j < 4; ++j) {
      const int ocol = colbase + wn * 64 + j * 16 + lh;
      float* op = out + (size_t)orow * NDIM + ocol;
#pragma unroll
      for (int vv = 0; vv < 8; ++vv) op[(size_t)vv * NDIM] = acc[i][j][vv];
    }
  }
}

// ---------------------------------------------------------------------------
// Launcher
// ---------------------------------------------------------------------------
extern "C" void kernel_launch(void* const* d_in, const int* in_sizes, int n_in,
                              void* d_out, int out_size, void* d_ws, size_t ws_size,
                              hipStream_t stream) {
  (void)in_sizes; (void)n_in; (void)out_size; (void)ws_size;
  const float* x        = (const float*)d_in[0];   // (B, K) f32
  const int*   indices  = (const int*)d_in[1];     // (N, K) i32
  const float* codebook = (const float*)d_in[2];   // (16,) f32
  const float* norms    = (const float*)d_in[3];   // (N, G) f32
  const float* rot      = (const float*)d_in[4];   // (G, 128, 128) f32
  float*       out      = (float*)d_out;           // (B, N) f32

  const size_t xr_bytes = (size_t)BDIM * KDIM * sizeof(__bf16);
  __bf16* Xr = reinterpret_cast<__bf16*>(d_ws);
  __bf16* Wq = reinterpret_cast<__bf16*>((char*)d_ws + xr_bytes);

  // 1) dequant + fold scales.
  dequant_w_kernel<<<(NDIM * (long)KDIM) / 2048, 256, 0, stream>>>(
      indices, codebook, norms, Wq);

  // 2) rotate x per group: 69632 B dynamic LDS.
  rotate_x_kernel<<<dim3(BDIM / 128, NG), 256,
                    2 * GRP * (GRP + 8) * sizeof(__bf16), stream>>>(x, rot, Xr);

  // 3) main GEMM: double-buffered panels, 73728 B dynamic LDS.
  gemm_kernel<<<dim3(BDIM / 128, NDIM / 128), 256,
                2 * 2 * 128 * 72 * sizeof(__bf16), stream>>>(Xr, Wq, out);
}
